// TrPredictor_29489245454795
// MI455X (gfx1250) — matrix operand order repaired and checked by reading, not compiled, not run on hardware
//
#include <hip/hip_runtime.h>

// ---- problem constants (from reference) ----
#define B_    4096
#define S_    201
#define E_    32
#define HID_  512
#define NTR_  4
#define NCAL_ 8
#define CW_   16
#define SP_   208   // S padded to 13 * 16 (M tiles)
#define KP_   224   // keys padded to 7 * 32 (WMMA K steps for A@V)
#define MT_   13    // number of 16-row M tiles
#define NW_   8     // waves per block (256 threads, wave32)
#define LN_EPS 1e-5f

typedef __attribute__((ext_vector_type(16))) _Float16 v16h;
typedef __attribute__((ext_vector_type(8)))  _Float16 v8h;
typedef __attribute__((ext_vector_type(8)))  float    v8f;

// ---- shared memory layout (bytes) ----
#define OFF_XRES   0        // [SP_][E_] f32              26624
#define OFF_ABUF   26624    // [SP_][E_] f16              13312
#define OFF_QBUF   39936    // [SP_][E_] f16              13312
#define OFF_KBUF   53248    // [SP_][E_] f16              13312
#define OFF_VBS    66560    // V swizzled: 14 blk * 512 h 14336
#define OFF_WQS    80896    // 2 blk * 512 h               2048
#define OFF_WKS    82944
#define OFF_WVS    84992
#define OFF_WOS    87040
#define OFF_W1S    89088    // 32 blk * 512 h             32768
#define OFF_W2S    121856   // 32 blk * 512 h             32768
#define OFF_SCORES 154624   // [NW_][16][KP_] f16         57344
#define OFF_CONTR  211968   // [SP_] f32                    832
#define OFF_WSF    212800   // async bounce buf f32       65536
#define SMEM_BYTES 278336   // < 320 KB WGP LDS

__device__ __forceinline__ v8f v8f_zero() {
  v8f z = {0.f,0.f,0.f,0.f,0.f,0.f,0.f,0.f};
  return z;
}

__device__ __forceinline__ v8f wmma16(v16h a, v16h b, v8f c) {
  // D = A(16x32 f16) * B(32x16 f16) + C(16x16 f32)
  return __builtin_amdgcn_wmma_f32_16x16x32_f16(false, a, false, b, (short)0, c,
                                                false, false);
}

// Async DMA: 16 bytes per lane, global -> LDS, tracked by ASYNCcnt.
// GV addressing mode: per-lane 64-bit global address, per-lane LDS byte address.
__device__ __forceinline__ void async_g2l_b128(unsigned lds_addr, const void* gaddr) {
  asm volatile("global_load_async_to_lds_b128 %0, %1, off"
               :: "v"(lds_addr), "v"(gaddr)
               : "memory");
}
__device__ __forceinline__ void wait_async0() {
  asm volatile("s_wait_asynccnt 0x0" ::: "memory");
}

// Swizzle index for pre-packed WMMA B operand (32x16 tile -> 512 halves):
// lane l = (N&15) + (K&16), half j = K&15; each lane's 16 halves are contiguous.
__device__ __forceinline__ int bswz(int K, int N) {
  return (((N & 15) + (K & 16)) << 4) + (K & 15);
}

// A-matrix 16x32 f16 loader (row-major src, 16B-aligned rows):
// lanes 0-15: row M=lane, K {0..15}; lanes 16-31: row M=lane-16, K {16..31}
// Both 8-half runs are contiguous -> two ds_load_b128.
__device__ __forceinline__ v16h load_a16(const _Float16* src, int ld) {
  int l = threadIdx.x & 31;
  const _Float16* row = src + (l & 15) * ld + ((l < 16) ? 0 : 16);
  v8h lo = *(const v8h*)(row);
  v8h hi = *(const v8h*)(row + 8);
  return __builtin_shufflevector(lo, hi, 0,1,2,3,4,5,6,7,8,9,10,11,12,13,14,15);
}

// Transposed B loader: logical B(K,N) = src[N*ld + K] (used for Q @ K^T).
// 16 contiguous halves per lane -> two ds_load_b128.
__device__ __forceinline__ v16h load_bt16(const _Float16* src, int ld) {
  int l = threadIdx.x & 31;
  const _Float16* row = src + (l & 15) * ld + ((l < 16) ? 0 : 16);
  v8h lo = *(const v8h*)(row);
  v8h hi = *(const v8h*)(row + 8);
  return __builtin_shufflevector(lo, hi, 0,1,2,3,4,5,6,7,8,9,10,11,12,13,14,15);
}

// Pre-swizzled B loader: blkbase points at a 512-half block in bswz layout.
__device__ __forceinline__ v16h load_bswz(const _Float16* blkbase) {
  const _Float16* p = blkbase + (threadIdx.x & 31) * 16;
  v8h lo = *(const v8h*)(p);
  v8h hi = *(const v8h*)(p + 8);
  return __builtin_shufflevector(lo, hi, 0,1,2,3,4,5,6,7,8,9,10,11,12,13,14,15);
}

__global__ __launch_bounds__(256) void trpred_kernel(
    const int* __restrict__ seq, const int* __restrict__ expid,
    const float* __restrict__ tok_emb, const float* __restrict__ pos_emb,
    const float* __restrict__ ln1_g, const float* __restrict__ ln1_b,
    const float* __restrict__ wq, const float* __restrict__ wk,
    const float* __restrict__ wv,
    const float* __restrict__ bq, const float* __restrict__ bk,
    const float* __restrict__ bv,
    const float* __restrict__ wo, const float* __restrict__ bo,
    const float* __restrict__ ln2_g, const float* __restrict__ ln2_b,
    const float* __restrict__ w1, const float* __restrict__ b1,
    const float* __restrict__ w2, const float* __restrict__ b2,
    const float* __restrict__ lnf_g, const float* __restrict__ lnf_b,
    const float* __restrict__ wpen, const float* __restrict__ bpen,
    const float* __restrict__ wfan, const float* __restrict__ bfan,
    const float* __restrict__ wcal, const float* __restrict__ bcal,
    float* __restrict__ out)
{
  extern __shared__ char smem[];
  float*    xres    = (float*)   (smem + OFF_XRES);
  _Float16* abuf    = (_Float16*)(smem + OFF_ABUF);
  _Float16* qbuf    = (_Float16*)(smem + OFF_QBUF);
  _Float16* kbuf    = (_Float16*)(smem + OFF_KBUF);
  _Float16* vbs     = (_Float16*)(smem + OFF_VBS);   // [kk*2+nt][512] bswz layout
  _Float16* wqs     = (_Float16*)(smem + OFF_WQS);
  _Float16* wks     = (_Float16*)(smem + OFF_WKS);
  _Float16* wvs     = (_Float16*)(smem + OFF_WVS);
  _Float16* wos     = (_Float16*)(smem + OFF_WOS);
  _Float16* w1s     = (_Float16*)(smem + OFF_W1S);   // [N/16][512]
  _Float16* w2s     = (_Float16*)(smem + OFF_W2S);   // [(K/32)*2 + N/16][512]
  _Float16* scores  = (_Float16*)(smem + OFF_SCORES);
  float*    contrib = (float*)   (smem + OFF_CONTR);
  float*    wsf     = (float*)   (smem + OFF_WSF);   // 16384-float async bounce

  const int b    = blockIdx.x;
  const int tid  = threadIdx.x;
  const int wave = tid >> 5;
  const int lane = tid & 31;
  const int n    = lane & 15;              // C/D layout: column within tile
  const int mb   = (lane < 16) ? 0 : 8;    // C/D layout: row base within tile
  _Float16* scw  = scores + wave * 16 * KP_;  // per-wave [16][KP_] staging

  // ---- embedding + positional add; padded rows zeroed ----
  for (int idx = tid; idx < SP_ * E_; idx += 256) {
    int s = idx >> 5, e = idx & 31;
    float v = 0.f;
    if (s < S_) {
      int t = seq[b * S_ + s];
      v = tok_emb[t * E_ + e] + pos_emb[s * E_ + e];
    }
    xres[idx] = v;
  }
  // zero V swizzle buffer once: the pad region (keys 208..223) is never written
  // by QKV, so it stays zero across all layers (prevents 0*NaN in A@V K-pad).
  for (int idx = tid; idx < 14 * 512; idx += 256) vbs[idx] = (_Float16)0.f;
  __syncthreads();

  for (int L = 0; L < NTR_; ++L) {
    // ---- kick off async DMA of w1 (f32, 64KB) into LDS bounce buffer ----
    {
      const float* nw1 = w1 + L * E_ * HID_;
#pragma unroll
      for (int i = 0; i < 16; ++i) {
        int idx = i * 1024 + tid * 4;     // 4 floats (16B) per lane per op
        async_g2l_b128((unsigned)(uintptr_t)(wsf + idx), nw1 + idx);
      }
    }
    // ---- overlap: stage small attention weights f32->f16 (pre-swizzled) ----
    for (int i = tid; i < E_ * E_; i += 256) {
      int K = i >> 5, N = i & 31;
      int d = (N >> 4) * 512 + bswz(K, N);
      wqs[d] = (_Float16)wq[L * E_ * E_ + i];
      wks[d] = (_Float16)wk[L * E_ * E_ + i];
      wvs[d] = (_Float16)wv[L * E_ * E_ + i];
      wos[d] = (_Float16)wo[L * E_ * E_ + i];
    }
    wait_async0();
    __syncthreads();
    // convert w1 bounce -> swizzled f16, then reuse bounce for w2
    for (int i = tid; i < E_ * HID_; i += 256) {
      int K = i >> 9, N = i & 511;              // w1: [32][512]
      w1s[(N >> 4) * 512 + bswz(K, N)] = (_Float16)wsf[i];
    }
    __syncthreads();
    {
      const float* nw2 = w2 + L * HID_ * E_;
#pragma unroll
      for (int i = 0; i < 16; ++i) {
        int idx = i * 1024 + tid * 4;
        async_g2l_b128((unsigned)(uintptr_t)(wsf + idx), nw2 + idx);
      }
    }
    wait_async0();
    __syncthreads();
    for (int i = tid; i < HID_ * E_; i += 256) {
      int K = i >> 5, N = i & 31;               // w2: [512][32]
      int blk = (K >> 5) * 2 + (N >> 4);
      w2s[blk * 512 + bswz(K & 31, N)] = (_Float16)wsf[i];
    }
    __syncthreads();

    // ---- LN1 -> abuf (one thread per row) ----
    if (tid < SP_) {
      const float* g  = ln1_g + L * E_;
      const float* be = ln1_b + L * E_;
      const float* row = xres + tid * E_;
      float m = 0.f;
#pragma unroll
      for (int e = 0; e < E_; ++e) m += row[e];
      m *= (1.f / E_);
      float var = 0.f;
#pragma unroll
      for (int e = 0; e < E_; ++e) { float d = row[e] - m; var += d * d; }
      float inv = rsqrtf(var * (1.f / E_) + LN_EPS);
#pragma unroll
      for (int e = 0; e < E_; ++e)
        abuf[tid * E_ + e] = (_Float16)((row[e] - m) * inv * g[e] + be[e]);
    }
    __syncthreads();

    // ---- Q,K,V projections (K = E = 32 -> one WMMA step) ----
    for (int mt = wave; mt < MT_; mt += NW_) {
      v16h a = load_a16(abuf + mt * 16 * E_, E_);
#pragma unroll
      for (int nt = 0; nt < 2; ++nt) {
        v8f cq = wmma16(a, load_bswz(wqs + nt * 512), v8f_zero());
        v8f ck = wmma16(a, load_bswz(wks + nt * 512), v8f_zero());
        v8f cv = wmma16(a, load_bswz(wvs + nt * 512), v8f_zero());
        float biq = bq[L * E_ + nt * 16 + n];
        float bik = bk[L * E_ + nt * 16 + n];
        float biv = bv[L * E_ + nt * 16 + n];
#pragma unroll
        for (int r = 0; r < 8; ++r) {
          int key = mt * 16 + mb + r;
          int off = key * E_ + nt * 16 + n;
          qbuf[off] = (_Float16)(cq[r] + biq);
          kbuf[off] = (_Float16)(ck[r] + bik);
          // V goes straight into swizzled B layout for the A@V pass
          int blk = (key >> 5) * 2 + nt;
          vbs[blk * 512 + bswz(key & 31, n)] = (_Float16)(cv[r] + biv);
        }
      }
    }
    __syncthreads();

    // ---- attention: scores, softmax, A@V, O-projection, residual ----
    for (int mt = wave; mt < MT_; mt += NW_) {
      v16h aq = load_a16(qbuf + mt * 16 * E_, E_);
      v8f sc[MT_];
#pragma unroll
      for (int kt = 0; kt < MT_; ++kt)
        sc[kt] = wmma16(aq, load_bt16(kbuf + kt * 16 * E_, E_), v8f_zero());

      const float scl = 0.17677669529663687f;  // 1/sqrt(E)
#pragma unroll
      for (int r = 0; r < 8; ++r) {
        float mx = -1e30f;
#pragma unroll
        for (int kt = 0; kt < MT_; ++kt) {
          float vv = sc[kt][r] * scl;
          if (kt * 16 + n >= S_) vv = -1e30f;   // mask padded keys
          sc[kt][r] = vv;
          mx = fmaxf(mx, vv);
        }
        mx = fmaxf(mx, __shfl_xor(mx, 1));
        mx = fmaxf(mx, __shfl_xor(mx, 2));
        mx = fmaxf(mx, __shfl_xor(mx, 4));
        mx = fmaxf(mx, __shfl_xor(mx, 8));
        float sum = 0.f;
#pragma unroll
        for (int kt = 0; kt < MT_; ++kt) {
          float ev = __expf(sc[kt][r] - mx);
          sc[kt][r] = ev;
          sum += ev;
        }
        sum += __shfl_xor(sum, 1);
        sum += __shfl_xor(sum, 2);
        sum += __shfl_xor(sum, 4);
        sum += __shfl_xor(sum, 8);
        float inv = 1.f / sum;
        int row = mb + r;
#pragma unroll
        for (int kt = 0; kt < MT_; ++kt)
          scw[row * KP_ + kt * 16 + n] = (_Float16)(sc[kt][r] * inv);
        scw[row * KP_ + 208 + n] = (_Float16)0.f;   // zero K-pad 208..223
      }

      // A @ V : 7 K-steps of 32, V pre-swizzled (pad blocks are zero)
      v8f o0 = v8f_zero(), o1 = v8f_zero();
#pragma unroll
      for (int kk = 0; kk < 7; ++kk) {
        v16h as = load_a16(scw + kk * 32, KP_);
        o0 = wmma16(as, load_bswz(vbs + (kk * 2 + 0) * 512), o0);
        o1 = wmma16(as, load_bswz(vbs + (kk * 2 + 1) * 512), o1);
      }

      // restage att tile f16 (C-layout -> row-major) via per-wave LDS;
      // same-wave DS ops are in order, no barrier needed.
#pragma unroll
      for (int r = 0; r < 8; ++r) {
        scw[(mb + r) * E_ + n]      = (_Float16)o0[r];
        scw[(mb + r) * E_ + 16 + n] = (_Float16)o1[r];
      }
      v16h aa = load_a16(scw, E_);
#pragma unroll
      for (int nt = 0; nt < 2; ++nt) {
        v8f c = wmma16(aa, load_bswz(wos + nt * 512), v8f_zero());
        float bi = bo[L * E_ + nt * 16 + n];
#pragma unroll
        for (int r = 0; r < 8; ++r)
          xres[(mt * 16 + mb + r) * E_ + nt * 16 + n] += c[r] + bi;
      }
    }
    __syncthreads();

    // ---- LN2 -> abuf ----
    if (tid < SP_) {
      const float* g  = ln2_g + L * E_;
      const float* be = ln2_b + L * E_;
      const float* row = xres + tid * E_;
      float m = 0.f;
#pragma unroll
      for (int e = 0; e < E_; ++e) m += row[e];
      m *= (1.f / E_);
      float var = 0.f;
#pragma unroll
      for (int e = 0; e < E_; ++e) { float d = row[e] - m; var += d * d; }
      float inv = rsqrtf(var * (1.f / E_) + LN_EPS);
#pragma unroll
      for (int e = 0; e < E_; ++e)
        abuf[tid * E_ + e] = (_Float16)((row[e] - m) * inv * g[e] + be[e]);
    }
    __syncthreads();

    // prefetch next layer's big weights into cache (global_prefetch_b8)
    if (L + 1 < NTR_) {
      const float* nw1 = w1 + (L + 1) * E_ * HID_;
      const float* nw2 = w2 + (L + 1) * HID_ * E_;
      for (int i = tid * 32; i < E_ * HID_; i += 256 * 32) {
        __builtin_prefetch(nw1 + i, 0, 1);
        __builtin_prefetch(nw2 + i, 0, 1);
      }
    }

    // ---- FFN, fused per 32-wide hidden chunk ----
    for (int mt = wave; mt < MT_; mt += NW_) {
      v16h a = load_a16(abuf + mt * 16 * E_, E_);
      v8f f0 = v8f_zero(), f1 = v8f_zero();
      for (int cc = 0; cc < HID_ / 32; ++cc) {
        v8f h0 = wmma16(a, load_bswz(w1s + (cc * 2 + 0) * 512), v8f_zero());
        v8f h1 = wmma16(a, load_bswz(w1s + (cc * 2 + 1) * 512), v8f_zero());
        float bb0 = b1[L * HID_ + cc * 32 + n];
        float bb1 = b1[L * HID_ + cc * 32 + 16 + n];
#pragma unroll
        for (int r = 0; r < 8; ++r) {
          float hv = h0[r] + bb0; hv = hv / (1.f + __expf(-hv));  // silu
          float hw = h1[r] + bb1; hw = hw / (1.f + __expf(-hw));
          scw[(mb + r) * E_ + n]      = (_Float16)hv;
          scw[(mb + r) * E_ + 16 + n] = (_Float16)hw;
        }
        v16h ah = load_a16(scw, E_);
        f0 = wmma16(ah, load_bswz(w2s + (cc * 2 + 0) * 512), f0);
        f1 = wmma16(ah, load_bswz(w2s + (cc * 2 + 1) * 512), f1);
      }
      float bba = b2[L * E_ + n];
      float bbb = b2[L * E_ + 16 + n];
#pragma unroll
      for (int r = 0; r < 8; ++r) {
        int row = mt * 16 + mb + r;
        xres[row * E_ + n]      += f0[r] + bba;
        xres[row * E_ + 16 + n] += f1[r] + bbb;
      }
    }
    __syncthreads();
    // re-zero padded rows so garbage can't turn into NaN and leak via 0*NaN
    for (int idx = tid; idx < (SP_ - S_) * E_; idx += 256) xres[S_ * E_ + idx] = 0.f;
    __syncthreads();
  }

  // ---- final head: LN_f, per-row pen dot, sum over S, fan/elu, calibrator ----
  if (tid < SP_) {
    float c = 0.f;
    if (tid < S_) {
      const float* row = xres + tid * E_;
      float m = 0.f;
#pragma unroll
      for (int e = 0; e < E_; ++e) m += row[e];
      m *= (1.f / E_);
      float var = 0.f;
#pragma unroll
      for (int e = 0; e < E_; ++e) { float d = row[e] - m; var += d * d; }
      float inv = rsqrtf(var * (1.f / E_) + LN_EPS);
      float dot = 0.f;
#pragma unroll
      for (int e = 0; e < E_; ++e)
        dot += ((row[e] - m) * inv * lnf_g[e] + lnf_b[e]) * wpen[e];
      c = dot + bpen[0];
    }
    contrib[tid] = c;
  }
  __syncthreads();
  if (tid == 0) {
    float u = 0.f;
    for (int s = 0; s < S_; ++s) u += contrib[s];
    int id = expid[b];
    float corr = bcal[id];
#pragma unroll
    for (int c = 0; c < CW_; ++c) {
      float p = u * wfan[c] + bfan[c];
      p = (p > 0.f) ? p : (__expf(p) - 1.f);  // elu
      corr += p * wcal[id * CW_ + c];
    }
    out[b] = corr + u;
  }
}

extern "C" void kernel_launch(void* const* d_in, const int* in_sizes, int n_in,
                              void* d_out, int out_size, void* d_ws, size_t ws_size,
                              hipStream_t stream) {
  const int*   seq     = (const int*)  d_in[0];
  const int*   expid   = (const int*)  d_in[1];
  const float* tok_emb = (const float*)d_in[2];
  const float* pos_emb = (const float*)d_in[3];
  const float* ln1_g   = (const float*)d_in[4];
  const float* ln1_b   = (const float*)d_in[5];
  const float* wq      = (const float*)d_in[6];
  const float* wk      = (const float*)d_in[7];
  const float* wv      = (const float*)d_in[8];
  const float* bq      = (const float*)d_in[9];
  const float* bk      = (const float*)d_in[10];
  const float* bv      = (const float*)d_in[11];
  const float* wo      = (const float*)d_in[12];
  const float* bo      = (const float*)d_in[13];
  const float* ln2_g   = (const float*)d_in[14];
  const float* ln2_b   = (const float*)d_in[15];
  const float* w1      = (const float*)d_in[16];
  const float* b1      = (const float*)d_in[17];
  const float* w2      = (const float*)d_in[18];
  const float* b2      = (const float*)d_in[19];
  const float* lnf_g   = (const float*)d_in[20];
  const float* lnf_b   = (const float*)d_in[21];
  const float* wpen    = (const float*)d_in[22];
  const float* bpen    = (const float*)d_in[23];
  const float* wfan    = (const float*)d_in[24];
  const float* bfan    = (const float*)d_in[25];
  const float* wcal    = (const float*)d_in[26];
  const float* bcal    = (const float*)d_in[27];

  trpred_kernel<<<dim3(B_), dim3(256), SMEM_BYTES, stream>>>(
      seq, expid, tok_emb, pos_emb, ln1_g, ln1_b,
      wq, wk, wv, bq, bk, bv, wo, bo, ln2_g, ln2_b,
      w1, b1, w2, b2, lnf_g, lnf_b, wpen, bpen, wfan, bfan, wcal, bcal,
      (float*)d_out);
}
